// GlobalAggregationBlock_91225105367612
// MI455X (gfx1250) — compile-verified, matrix operand order checked by reading, not run on hardware
//
#include <hip/hip_runtime.h>
#include <hip/hip_bf16.h>

// ---------------------------------------------------------------------------
// CDNA5 (gfx1250) wave32 WMMA implementation of the global-aggregation
// (1x1-conv self-attention) block.  All heavy math goes through
// v_wmma_f32_16x16x32_bf16; softmax is flash-attention style so the
// 4096x4096 attention matrix never touches memory.  K/V working set
// (16 MB bf16) lives in the 192 MB L2.
// ---------------------------------------------------------------------------

typedef __attribute__((ext_vector_type(16))) __bf16 v16bf;
typedef __attribute__((ext_vector_type(8)))  float  v8f;
typedef __attribute__((ext_vector_type(8)))  unsigned int v8u;

union frag_u { v16bf bf; v8u u; };

#define B_  8
#define C_  256
#define N_  4096
#define CK_ 128
#define CV_ 128
#define CO_ 256

// ---------------------------------------------------------------------------
// Fragment loader for 16-bit A/B operands of v_wmma_*_16x16x32.
// ISA 7.12.2: lane L holds row/col (L&15); lanes 0-15 carry K 0..7,16..23,
// lanes 16-31 K 8..15,24..31; VGPR p holds an even K-pair:
//   p<4 : K = 2p        + 8*(L>>4)
//   p>=4: K = 16+2(p-4) + 8*(L>>4)
// All K-pairs are contiguous even offsets -> aligned u32 loads, which the
// compiler merges into paired global_load_b128 per lane.
// ---------------------------------------------------------------------------
__device__ __forceinline__ v16bf load_frag(const __bf16* base, int ld, int lane) {
    frag_u f;
    const unsigned int* row =
        (const unsigned int*)(base + (size_t)(lane & 15) * (size_t)ld);
    const int half8 = (lane >> 4) << 3;
#pragma unroll
    for (int p = 0; p < 8; ++p) {
        const int k = ((p < 4) ? (2 * p) : (16 + 2 * (p - 4))) + half8;
        f.u[p] = row[k >> 1];
    }
    return f.bf;
}

__device__ __forceinline__ v8f wmma_bf16(v16bf a, v16bf b, v8f c) {
    // args: neg_a, A, neg_b, B, c_mod, C, reuse_a, reuse_b
    return __builtin_amdgcn_wmma_f32_16x16x32_bf16(false, a, false, b,
                                                   (short)0, c, false, false);
}

// ---------------------------------------------------------------------------
// Small prep kernels
// ---------------------------------------------------------------------------
__global__ void cvt_f32_bf16_kernel(const float* __restrict__ in,
                                    __bf16* __restrict__ out, int n) {
    int i = blockIdx.x * 256 + threadIdx.x;
    if (i < n) out[i] = (__bf16)in[i];
}

// x (B, C, N) f32  ->  xt (B, N, C) bf16      (LDS 32x32 tile transpose)
__global__ void transpose_x_kernel(const float* __restrict__ x,
                                   __bf16* __restrict__ xt) {
    __shared__ float tile[32][33];
    const int b  = blockIdx.z;
    const int n0 = blockIdx.x * 32;
    const int c0 = blockIdx.y * 32;
    const float* xb = x + (size_t)b * C_ * N_;
#pragma unroll
    for (int i = threadIdx.y; i < 32; i += 8)
        tile[i][threadIdx.x] = xb[(size_t)(c0 + i) * N_ + n0 + threadIdx.x];
    __syncthreads();
    __bf16* xtb = xt + (size_t)b * N_ * C_;
#pragma unroll
    for (int i = threadIdx.y; i < 32; i += 8)
        xtb[(size_t)(n0 + i) * C_ + c0 + threadIdx.x] =
            (__bf16)tile[threadIdx.x][i];
}

// u viewed as (128, 4096) bf16 per batch -> ut (4096, 128) bf16 per batch.
// This implements the reference's *raw* reshape of (N, cv) into (cv, H, W).
__global__ void transpose_u_kernel(const __bf16* __restrict__ u,
                                   __bf16* __restrict__ ut) {
    __shared__ __bf16 tile[32][33];
    const int b  = blockIdx.z;
    const int t0 = blockIdx.x * 32;
    const int v0 = blockIdx.y * 32;
    const __bf16* ub = u + (size_t)b * CV_ * N_;
#pragma unroll
    for (int i = threadIdx.y; i < 32; i += 8)
        tile[i][threadIdx.x] = ub[(size_t)(v0 + i) * N_ + t0 + threadIdx.x];
    __syncthreads();
    __bf16* utb = ut + (size_t)b * N_ * CV_;
#pragma unroll
    for (int i = threadIdx.y; i < 32; i += 8)
        utb[(size_t)(t0 + i) * CV_ + v0 + threadIdx.x] = tile[threadIdx.x][i];
}

// ---------------------------------------------------------------------------
// Stage 1: QKV projection.  Per wave: 16 tokens, K = C = 256.
//   q,k stored token-major (B, N, 128) bf16 via LDS repack -> b128 stores
//   v   stored channel-major (B, 128, N) bf16 (packed 16B stores from D-tiles)
// ---------------------------------------------------------------------------
__global__ __launch_bounds__(256) void qkv_kernel(
    const __bf16* __restrict__ xt,
    const __bf16* __restrict__ Wqb, const __bf16* __restrict__ Wkb,
    const __bf16* __restrict__ Wvb,
    const float* __restrict__ bq, const float* __restrict__ bk,
    const float* __restrict__ bv,
    __bf16* __restrict__ qb, __bf16* __restrict__ kb,
    __bf16* __restrict__ vb) {
    __shared__ __bf16 ldsT[8][16 * 16];        // per-wave 512B repack tile
    const int b    = blockIdx.y;
    const int wave = threadIdx.x >> 5;
    const int lane = threadIdx.x & 31;
    const int half = lane >> 4;
    const int nl   = lane & 15;
    const int tok0 = blockIdx.x * 128 + wave * 16;
    __bf16* myT = ldsT[wave];

    const __bf16* xtb = xt + (size_t)b * N_ * C_ + (size_t)tok0 * C_;
    v16bf a[8];
#pragma unroll
    for (int c = 0; c < 8; ++c) a[c] = load_frag(xtb + c * 32, C_, lane);

    const __bf16* Ws[3] = {Wqb, Wkb, Wvb};
    const float*  bs[3] = {bq, bk, bv};

#pragma unroll
    for (int m = 0; m < 3; ++m) {
#pragma unroll
        for (int t = 0; t < 8; ++t) {          // output-channel tile
            // batch all 8 weight fragments first: 16 b128 loads in flight
            v16bf wf[8];
#pragma unroll
            for (int c = 0; c < 8; ++c)
                wf[c] = load_frag(Ws[m] + (size_t)(t * 16) * C_ + c * 32,
                                  C_, lane);
            v8f acc = {};
#pragma unroll
            for (int c = 0; c < 8; ++c) acc = wmma_bf16(a[c], wf[c], acc);

            const int   col  = t * 16 + nl;
            const float bias = bs[m][col];
            if (m < 2) {
                // repack C-tile through LDS -> one b128 store per lane
#pragma unroll
                for (int r = 0; r < 8; ++r)
                    myT[(r + 8 * half) * 16 + nl] = (__bf16)(acc[r] + bias);
                asm volatile("s_wait_dscnt 0" ::: "memory");
                const uint4 val = ((const uint4*)myT)[lane];  // row=lane>>1
                __bf16* dst = (m == 0 ? qb : kb) + (size_t)b * N_ * CK_ +
                              (size_t)(tok0 + (lane >> 1)) * CK_ +
                              t * 16 + (lane & 1) * 8;
                *(uint4*)dst = val;
                asm volatile("" ::: "memory");
            } else {
                union { __bf16 h[8]; uint4 u; } pk;
#pragma unroll
                for (int r = 0; r < 8; ++r) pk.h[r] = (__bf16)(acc[r] + bias);
                __bf16* dst = vb + (size_t)b * CV_ * N_ +
                              (size_t)col * N_ + tok0 + 8 * half;
                *(uint4*)dst = pk.u;           // 8 consecutive tokens
            }
        }
    }
}

// ---------------------------------------------------------------------------
// Stage 2: flash attention.  Per wave: 16 query rows x all 4096 keys in
// 32-key chunks.  Software-pipelined: next chunk's K fragments reload in
// place right after the S WMMAs, V fragments are issued before the softmax
// block (the s_wait_dscnt memory clobber keeps them there), so exp/shuffle
// math hides the global-load latency.
// ---------------------------------------------------------------------------
__global__ __launch_bounds__(256) void flash_kernel(
    const __bf16* __restrict__ qb, const __bf16* __restrict__ kb,
    const __bf16* __restrict__ vb, __bf16* __restrict__ ob) {
    __shared__ __bf16 ldsP[8][16 * 32];        // 1 KB/wave: P re-layout
    __shared__ __bf16 ldsO[8][16 * 128];       // 4 KB/wave: output repack
    const int b    = blockIdx.y;
    const int wave = threadIdx.x >> 5;
    const int lane = threadIdx.x & 31;
    const int half = lane >> 4;
    const int nl   = lane & 15;
    const int row0 = blockIdx.x * 128 + wave * 16;

    const __bf16* qbb = qb + (size_t)b * N_ * CK_ + (size_t)row0 * CK_;
    const __bf16* kbb = kb + (size_t)b * N_ * CK_;
    const __bf16* vbb = vb + (size_t)b * CV_ * N_;
    __bf16* myP = ldsP[wave];

    v16bf aq[4];
#pragma unroll
    for (int c = 0; c < 4; ++c) aq[c] = load_frag(qbb + c * 32, CK_, lane);

    v8f o[8];
#pragma unroll
    for (int t = 0; t < 8; ++t) { v8f z = {}; o[t] = z; }
    float mst[8], lst[8];
#pragma unroll
    for (int r = 0; r < 8; ++r) { mst[r] = -INFINITY; lst[r] = 0.0f; }
    const float scale = 0.08838834764831845f;  // 1/sqrt(128)

    // prologue: K fragments for chunk 0
    v16bf kf[8];
#pragma unroll
    for (int c = 0; c < 4; ++c) {
        kf[c]     = load_frag(kbb + c * 32, CK_, lane);
        kf[4 + c] = load_frag(kbb + (size_t)16 * CK_ + c * 32, CK_, lane);
    }

    for (int m0 = 0; m0 < N_; m0 += 32) {
        v8f s0 = {}, s1 = {};
#pragma unroll
        for (int c = 0; c < 4; ++c) {
            s0 = wmma_bf16(aq[c], kf[c], s0);
            s1 = wmma_bf16(aq[c], kf[4 + c], s1);
        }
        // pipeline: reload kf with next chunk (WAR resolved by regalloc)
        const int mn = m0 + 32;
        if (mn < N_) {
#pragma unroll
            for (int c = 0; c < 4; ++c) {
                kf[c]     = load_frag(kbb + (size_t)mn * CK_ + c * 32,
                                      CK_, lane);
                kf[4 + c] = load_frag(kbb + (size_t)(mn + 16) * CK_ + c * 32,
                                      CK_, lane);
            }
        }
        // V fragments for the current chunk: issue before the softmax math
        v16bf vf[8];
#pragma unroll
        for (int t = 0; t < 8; ++t)
            vf[t] = load_frag(vbb + (size_t)(t * 16) * N_ + m0, N_, lane);
        // stream prefetch two chunks ahead (global_prefetch_b8)
        if (m0 + 64 < N_) {
            __builtin_prefetch(kbb + (size_t)(m0 + 64) * CK_, 0, 1);
            __builtin_prefetch(vbb + (size_t)(nl * 16) * N_ + m0 + 64, 0, 1);
        }

        float alpha[8];
#pragma unroll
        for (int r = 0; r < 8; ++r) {
            float x0 = s0[r] * scale;
            float x1 = s1[r] * scale;
            float mt = fmaxf(x0, x1);
            // row-max over the 16 lanes of this half (D-layout: N = lane&15)
            mt = fmaxf(mt, __shfl_xor(mt, 1, 32));
            mt = fmaxf(mt, __shfl_xor(mt, 2, 32));
            mt = fmaxf(mt, __shfl_xor(mt, 4, 32));
            mt = fmaxf(mt, __shfl_xor(mt, 8, 32));
            const float mn2 = fmaxf(mst[r], mt);
            const float al  = __expf(mst[r] - mn2);
            const float p0  = __expf(x0 - mn2);
            const float p1  = __expf(x1 - mn2);
            float ps = p0 + p1;
            ps += __shfl_xor(ps, 1, 32);
            ps += __shfl_xor(ps, 2, 32);
            ps += __shfl_xor(ps, 4, 32);
            ps += __shfl_xor(ps, 8, 32);
            lst[r]   = lst[r] * al + ps;
            mst[r]   = mn2;
            alpha[r] = al;
            // stage P (bf16) in row-major 16x32 for the A-fragment reload
            myP[(r + 8 * half) * 32 + nl]      = (__bf16)p0;
            myP[(r + 8 * half) * 32 + 16 + nl] = (__bf16)p1;
        }
#pragma unroll
        for (int t = 0; t < 8; ++t)
#pragma unroll
            for (int r = 0; r < 8; ++r) o[t][r] *= alpha[r];

        // DS ops are in-order per wave; wait for the b16 stores to land.
        asm volatile("s_wait_dscnt 0" ::: "memory");
        v16bf ap = load_frag(myP, 32, lane);
#pragma unroll
        for (int t = 0; t < 8; ++t) o[t] = wmma_bf16(ap, vf[t], o[t]);
    }

    // finalize: repack the 16x128 row block through LDS, coalesced stores
    float invl[8];
#pragma unroll
    for (int r = 0; r < 8; ++r) invl[r] = 1.0f / lst[r];
    __bf16* myO = ldsO[wave];
#pragma unroll
    for (int t = 0; t < 8; ++t)
#pragma unroll
        for (int r = 0; r < 8; ++r)
            myO[(r + 8 * half) * 128 + t * 16 + nl] =
                (__bf16)(o[t][r] * invl[r]);
    asm volatile("s_wait_dscnt 0" ::: "memory");
    const uint4* src = (const uint4*)myO;
    uint4* dst = (uint4*)(ob + (size_t)b * N_ * CV_ + (size_t)row0 * CV_);
#pragma unroll
    for (int j = 0; j < 8; ++j) dst[lane + 32 * j] = src[lane + 32 * j];
}

// ---------------------------------------------------------------------------
// Stage 3: output projection  y = Wo @ U + bo  with U the raw-reshaped
// attention output.  A = Ut (token-major), B = Wo (row-major = contiguous K).
// ---------------------------------------------------------------------------
__global__ __launch_bounds__(256) void out_proj_kernel(
    const __bf16* __restrict__ ut, const __bf16* __restrict__ Wob,
    const float* __restrict__ bo, float* __restrict__ y) {
    const int b    = blockIdx.y;
    const int wave = threadIdx.x >> 5;
    const int lane = threadIdx.x & 31;
    const int half = lane >> 4;
    const int nl   = lane & 15;
    const int t0   = blockIdx.x * 128 + wave * 16;

    const __bf16* utb = ut + (size_t)b * N_ * CV_ + (size_t)t0 * CV_;
    v16bf a[4];
#pragma unroll
    for (int c = 0; c < 4; ++c) a[c] = load_frag(utb + c * 32, CV_, lane);

    float* yb = y + (size_t)b * CO_ * N_;
#pragma unroll
    for (int ot = 0; ot < 16; ++ot) {
        v16bf wf[4];
#pragma unroll
        for (int c = 0; c < 4; ++c)
            wf[c] = load_frag(Wob + (size_t)(ot * 16) * CV_ + c * 32,
                              CV_, lane);
        v8f acc = {};
#pragma unroll
        for (int c = 0; c < 4; ++c) acc = wmma_bf16(a[c], wf[c], acc);

        const int   och  = ot * 16 + nl;
        const float bias = bo[och];
        float* dst = yb + (size_t)och * N_ + t0 + 8 * half;
        float4 lo = {acc[0] + bias, acc[1] + bias, acc[2] + bias, acc[3] + bias};
        float4 hi = {acc[4] + bias, acc[5] + bias, acc[6] + bias, acc[7] + bias};
        *(float4*)dst       = lo;
        *((float4*)dst + 1) = hi;
    }
}

// ---------------------------------------------------------------------------
// Launcher
// ---------------------------------------------------------------------------
extern "C" void kernel_launch(void* const* d_in, const int* in_sizes, int n_in,
                              void* d_out, int out_size, void* d_ws,
                              size_t ws_size, hipStream_t stream) {
    (void)in_sizes; (void)n_in; (void)out_size; (void)ws_size;
    const float* x  = (const float*)d_in[0];
    const float* Wq = (const float*)d_in[1];
    const float* bq = (const float*)d_in[2];
    const float* Wk = (const float*)d_in[3];
    const float* bk = (const float*)d_in[4];
    const float* Wv = (const float*)d_in[5];
    const float* bv = (const float*)d_in[6];
    const float* Wo = (const float*)d_in[7];
    const float* bo = (const float*)d_in[8];

    char*  w   = (char*)d_ws;
    size_t off = 0;
    auto alloc = [&](size_t bytes) -> void* {
        void* p = (void*)(w + off);
        off += (bytes + 255) & ~(size_t)255;
        return p;
    };
    __bf16* xt  = (__bf16*)alloc((size_t)B_ * N_ * C_ * 2);   // 16 MB
    __bf16* qb  = (__bf16*)alloc((size_t)B_ * N_ * CK_ * 2);  //  8 MB
    __bf16* kb  = (__bf16*)alloc((size_t)B_ * N_ * CK_ * 2);  //  8 MB
    __bf16* vb  = (__bf16*)alloc((size_t)B_ * CV_ * N_ * 2);  //  8 MB
    __bf16* ob  = (__bf16*)alloc((size_t)B_ * N_ * CV_ * 2);  //  8 MB
    __bf16* ut  = (__bf16*)alloc((size_t)B_ * N_ * CV_ * 2);  //  8 MB
    __bf16* Wqb = (__bf16*)alloc((size_t)CK_ * C_ * 2);
    __bf16* Wkb = (__bf16*)alloc((size_t)CK_ * C_ * 2);
    __bf16* Wvb = (__bf16*)alloc((size_t)CV_ * C_ * 2);
    __bf16* Wob = (__bf16*)alloc((size_t)CO_ * CV_ * 2);

    const int nw = CK_ * C_;  // 32768 for all four weights
    cvt_f32_bf16_kernel<<<(nw + 255) / 256, 256, 0, stream>>>(Wq, Wqb, nw);
    cvt_f32_bf16_kernel<<<(nw + 255) / 256, 256, 0, stream>>>(Wk, Wkb, nw);
    cvt_f32_bf16_kernel<<<(nw + 255) / 256, 256, 0, stream>>>(Wv, Wvb, nw);
    cvt_f32_bf16_kernel<<<(nw + 255) / 256, 256, 0, stream>>>(Wo, Wob, nw);

    transpose_x_kernel<<<dim3(N_ / 32, C_ / 32, B_), dim3(32, 8), 0, stream>>>(
        x, xt);

    qkv_kernel<<<dim3(N_ / 128, B_), 256, 0, stream>>>(
        xt, Wqb, Wkb, Wvb, bq, bk, bv, qb, kb, vb);

    flash_kernel<<<dim3(N_ / 128, B_), 256, 0, stream>>>(qb, kb, vb, ob);

    transpose_u_kernel<<<dim3(N_ / 32, CV_ / 32, B_), dim3(32, 8), 0,
                         stream>>>(ob, ut);

    out_proj_kernel<<<dim3(N_ / 128, B_), 256, 0, stream>>>(
        ut, Wob, bo, (float*)d_out);
}
